// BatchCriterion_83081847374373
// MI455X (gfx1250) — matrix operand order repaired and checked by reading, not compiled, not run on hardware
//
#include <hip/hip_runtime.h>
#include <hip/hip_bf16.h>

typedef _Float16 v4h  __attribute__((ext_vector_type(4)));
typedef _Float16 v8h  __attribute__((ext_vector_type(8)));
typedef _Float16 v16h __attribute__((ext_vector_type(16)));
typedef float    v8f  __attribute__((ext_vector_type(8)));

#define DIMS 128
#define INV_T 10.0f
#define NEG_M_C 1.0f
// exp(x * INV_T) == exp2(x * INV_T * log2(e)) -- one multiply feeds v_exp_f32
#define EXP2_SCALE 14.426950408889634f
#define LN2F 0.6931471805599453f

__device__ __forceinline__ const float* row_base(const float* f1, const float* f2,
                                                 int row, int half) {
    return (row < half) ? (f1 + (size_t)row * DIMS)
                        : (f2 + (size_t)(row - half) * DIMS);
}

__global__ __launch_bounds__(256) void nce_zero(float* all_div, float* acc, int N) {
    int i = blockIdx.x * blockDim.x + threadIdx.x;
    if (i < N) all_div[i] = 0.0f;
    if (i < 8) acc[i] = 0.0f;
}

// One-time f32 -> f16 conversion of x = concat(f1, f2) into workspace (2 MB,
// resident in the 192 MB L2 for both GEMM passes).
__global__ __launch_bounds__(256) void nce_cvt(const float* __restrict__ f1,
                                               const float* __restrict__ f2,
                                               _Float16* __restrict__ xh, int half) {
    int idx4 = blockIdx.x * blockDim.x + threadIdx.x;
    int halfElems = half * DIMS;                 // multiple of 4: no straddle
    int total4 = (2 * halfElems) >> 2;
    if (idx4 >= total4) return;
    int e = idx4 << 2;
    const float* src = (e < halfElems) ? (f1 + e) : (f2 + (e - halfElems));
    float4 v = *(const float4*)src;
    v4h o = {(_Float16)v.x, (_Float16)v.y, (_Float16)v.z, (_Float16)v.w};
    *(v4h*)(xh + e) = o;
}

// pos[i] = exp(dot(x_i, x_{(i+half)%N}) / T)  -- exact f32 dot
__global__ __launch_bounds__(256) void nce_pos(const float* __restrict__ f1,
                                               const float* __restrict__ f2,
                                               float* __restrict__ pos, int half) {
    int N = 2 * half;
    int i = blockIdx.x * blockDim.x + threadIdx.x;
    if (i >= N) return;
    const float* xi = row_base(f1, f2, i, half);
    int j = i + half; if (j >= N) j -= N;
    const float* xj = row_base(f1, f2, j, half);
    float dot = 0.0f;
#pragma unroll
    for (int k = 0; k < DIMS; k += 4) {
        float4 a = *(const float4*)(xi + k);
        float4 b = *(const float4*)(xj + k);
        dot += a.x * b.x + a.y * b.y + a.z * b.z + a.w * b.w;
    }
    pos[i] = __builtin_amdgcn_exp2f(dot * EXP2_SCALE);   // arg bounded, raw exp2 safe
}

// A fragment (16-bit A 16x32 layout): per lane M = lane%16, h = lane>>4,
//   elements 0..7  = K 8h .. 8h+7      (one b128)
//   elements 8..15 = K 16+8h .. 23+8h  (one b128)
__device__ __forceinline__ v16h load_afrag(const _Float16* row, int f, int h) {
    v8h lo = *(const v8h*)(row + 32 * f + 8 * h);
    v8h hi = *(const v8h*)(row + 32 * f + 16 + 8 * h);
    return __builtin_shufflevector(lo, hi, 0, 1, 2, 3, 4, 5, 6, 7,
                                   8, 9, 10, 11, 12, 13, 14, 15);
}

// B fragment (32x16 layout): per lane N = lane%16, h = lane>>4,
//   elements 0..7  = K 8h .. 8h+7      (one b128)
//   elements 8..15 = K 16+8h .. 23+8h  (one b128)
__device__ __forceinline__ v16h load_bfrag(const _Float16* row, int f, int h) {
    v8h lo = *(const v8h*)(row + 32 * f + 8 * h);
    v8h hi = *(const v8h*)(row + 32 * f + 16 + 8 * h);
    return __builtin_shufflevector(lo, hi, 0, 1, 2, 3, 4, 5, 6, 7,
                                   8, 9, 10, 11, 12, 13, 14, 15);
}

// Compute one 16x16 sim tile into c (f32) against the preloaded A panel.
// All B b128 loads issue in clauses before the 4 WMMAs drain them progressively.
__device__ __forceinline__ v8f tile_gemm(const v16h afrag[4], const _Float16* brow, int h) {
    v16h b0 = load_bfrag(brow, 0, h);
    v16h b1 = load_bfrag(brow, 1, h);
    v16h b2 = load_bfrag(brow, 2, h);
    v16h b3 = load_bfrag(brow, 3, h);
    v8f c = {};
    c = __builtin_amdgcn_wmma_f32_16x16x32_f16(false, afrag[0], false, b0, (short)0, c, false, false);
    c = __builtin_amdgcn_wmma_f32_16x16x32_f16(false, afrag[1], false, b1, (short)0, c, false, false);
    c = __builtin_amdgcn_wmma_f32_16x16x32_f16(false, afrag[2], false, b2, (short)0, c, false, false);
    c = __builtin_amdgcn_wmma_f32_16x16x32_f16(false, afrag[3], false, b3, (short)0, c, false, false);
    return c;
}

// Pass A: all_div[i] = sum_{j != i} exp(sim_ij / T)
__global__ __launch_bounds__(256) void nce_rowsum(const _Float16* __restrict__ xh,
                                                  float* __restrict__ all_div, int N) {
    const int rowTile = blockIdx.x << 4;
    const int wave = threadIdx.x >> 5;
    const int lane = threadIdx.x & 31;
    const int mLane = lane & 15;
    const int hi = lane >> 4;

    const _Float16* arow = xh + (size_t)(rowTile + mLane) * DIMS;
    v16h afrag[4];
#pragma unroll
    for (int f = 0; f < 4; ++f) afrag[f] = load_afrag(arow, f, hi);

    bool laneDiag[8];
#pragma unroll
    for (int r = 0; r < 8; ++r) laneDiag[r] = (mLane == r + (hi << 3));

    v8f sum = {0.f, 0.f, 0.f, 0.f, 0.f, 0.f, 0.f, 0.f};
    const int nColTiles = N >> 4;
    const _Float16* brow = xh + (size_t)((wave << 4) + mLane) * DIMS;
    for (int jt = wave; jt < nColTiles; jt += 8, brow += (size_t)8 * 16 * DIMS) {
        v8f c = tile_gemm(afrag, brow, hi);
        v8f e;
#pragma unroll
        for (int r = 0; r < 8; ++r)
            e[r] = __builtin_amdgcn_exp2f(c[r] * EXP2_SCALE);  // raw v_exp_f32
        if (jt == (int)blockIdx.x) {                           // uniform branch
#pragma unroll
            for (int r = 0; r < 8; ++r)
                if (laneDiag[r]) e[r] = 0.0f;                  // mask diagonal
        }
#pragma unroll
        for (int r = 0; r < 8; ++r) sum[r] += e[r];
    }
#pragma unroll
    for (int r = 0; r < 8; ++r) {
        float v = sum[r];
        v += __shfl_xor(v, 1, 32);
        v += __shfl_xor(v, 2, 32);
        v += __shfl_xor(v, 4, 32);
        v += __shfl_xor(v, 8, 32);
        if (mLane == 0)
            atomicAdd(&all_div[rowTile + r + (hi << 3)], v);
    }
}

// Pass B: acc[0] += sum_{i, j != i} log1p(-exp(sim_ij/T) / all_div[i])
// p <= ~1e-3 here, so log1p(-p) == log(1-p); accumulate log2(1-p) and scale
// the wave total by ln2 once at the end.
__global__ __launch_bounds__(256) void nce_logsum(const _Float16* __restrict__ xh,
                                                  const float* __restrict__ all_div,
                                                  float* __restrict__ acc, int N) {
    const int rowTile = blockIdx.x << 4;
    const int wave = threadIdx.x >> 5;
    const int lane = threadIdx.x & 31;
    const int mLane = lane & 15;
    const int hi = lane >> 4;

    const _Float16* arow = xh + (size_t)(rowTile + mLane) * DIMS;
    v16h afrag[4];
#pragma unroll
    for (int f = 0; f < 4; ++f) afrag[f] = load_afrag(arow, f, hi);

    bool laneDiag[8];
    float invDiv[8];
#pragma unroll
    for (int r = 0; r < 8; ++r) {
        laneDiag[r] = (mLane == r + (hi << 3));
        invDiv[r] = 1.0f / all_div[rowTile + r + (hi << 3)];
    }

    v8f sum = {0.f, 0.f, 0.f, 0.f, 0.f, 0.f, 0.f, 0.f};
    const int nColTiles = N >> 4;
    const _Float16* brow = xh + (size_t)((wave << 4) + mLane) * DIMS;
    for (int jt = wave; jt < nColTiles; jt += 8, brow += (size_t)8 * 16 * DIMS) {
        v8f c = tile_gemm(afrag, brow, hi);
        v8f e;
#pragma unroll
        for (int r = 0; r < 8; ++r)
            e[r] = __builtin_amdgcn_exp2f(c[r] * EXP2_SCALE);
        if (jt == (int)blockIdx.x) {                           // uniform branch
#pragma unroll
            for (int r = 0; r < 8; ++r)
                if (laneDiag[r]) e[r] = 0.0f;                  // diag -> log2(1) == 0
        }
#pragma unroll
        for (int r = 0; r < 8; ++r)
            sum[r] += __builtin_amdgcn_logf(__builtin_fmaf(-e[r], invDiv[r], 1.0f));
    }
#pragma unroll
    for (int r = 0; r < 8; ++r) {
        float v = sum[r];
        v += __shfl_xor(v, 1, 32);
        v += __shfl_xor(v, 2, 32);
        v += __shfl_xor(v, 4, 32);
        v += __shfl_xor(v, 8, 32);
        sum[r] = v;
    }
    if (mLane == 0) {
        float t = 0.0f;
#pragma unroll
        for (int r = 0; r < 8; ++r) t += sum[r];
        atomicAdd(&acc[0], t * LN2F);                          // log2 -> ln, once
    }
}

// Per-row closing terms: acc[1] += log(pos_i) - log(all_div_i);
//                        acc[2] += log1p(-pos_i/all_div_i)
__global__ __launch_bounds__(256) void nce_finalize(const float* __restrict__ pos,
                                                    const float* __restrict__ all_div,
                                                    float* __restrict__ acc, int N) {
    __shared__ float s1[256];
    __shared__ float s2[256];
    int i = blockIdx.x * blockDim.x + threadIdx.x;
    float t1 = 0.0f, t2 = 0.0f;
    if (i < N) {
        float p = pos[i];
        float d = all_div[i];
        t1 = (__builtin_amdgcn_logf(p) - __builtin_amdgcn_logf(d)) * LN2F;
        t2 = log1pf(-p / d);
    }
    s1[threadIdx.x] = t1;
    s2[threadIdx.x] = t2;
    __syncthreads();
    for (int s = 128; s > 0; s >>= 1) {
        if ((int)threadIdx.x < s) {
            s1[threadIdx.x] += s1[threadIdx.x + s];
            s2[threadIdx.x] += s2[threadIdx.x + s];
        }
        __syncthreads();
    }
    if (threadIdx.x == 0) {
        atomicAdd(&acc[1], s1[0]);
        atomicAdd(&acc[2], s2[0]);
    }
}

__global__ void nce_combine(const float* __restrict__ acc, float* __restrict__ out, int N) {
    float lnPonsum = (acc[0] - acc[2]) * NEG_M_C;  // full log1p sum minus pos-pair terms
    float lnPmtsum = acc[1];
    out[0] = -(lnPmtsum + lnPonsum) / (float)N;
}

extern "C" void kernel_launch(void* const* d_in, const int* in_sizes, int n_in,
                              void* d_out, int out_size, void* d_ws, size_t ws_size,
                              hipStream_t stream) {
    const float* f1 = (const float*)d_in[0];
    const float* f2 = (const float*)d_in[1];
    (void)n_in; (void)out_size; (void)ws_size;

    const int half = in_sizes[0] / DIMS;   // 4096
    const int N = 2 * half;                // 8192

    // ws layout (floats): all_div[N] | pos[N] | acc[8] | xh (N*DIMS f16, 16B-aligned)
    float* ws      = (float*)d_ws;
    float* all_div = ws;
    float* pos     = ws + N;
    float* acc     = ws + 2 * N;
    _Float16* xh   = (_Float16*)(ws + 2 * N + 8);

    float* out = (float*)d_out;

    const int tpb = 256;
    const int cvt4 = (N * DIMS) / 4;
    nce_zero    <<<(N + tpb - 1) / tpb, tpb, 0, stream>>>(all_div, acc, N);
    nce_cvt     <<<(cvt4 + tpb - 1) / tpb, tpb, 0, stream>>>(f1, f2, xh, half);
    nce_pos     <<<(N + tpb - 1) / tpb, tpb, 0, stream>>>(f1, f2, pos, half);
    nce_rowsum  <<<N / 16, tpb, 0, stream>>>(xh, all_div, N);
    nce_logsum  <<<N / 16, tpb, 0, stream>>>(xh, all_div, acc, N);
    nce_finalize<<<(N + tpb - 1) / tpb, tpb, 0, stream>>>(pos, all_div, acc, N);
    nce_combine <<<1, 1, 0, stream>>>(acc, out, N);
}
